// RoPEAttentionLayer_65103114273198
// MI455X (gfx1250) — matrix operand order repaired and checked by reading, not compiled, not run on hardware
//
#include <hip/hip_runtime.h>

// ---------------- CDNA5 WMMA types ----------------
typedef __attribute__((ext_vector_type(16))) __bf16 v16bf;
typedef __attribute__((ext_vector_type(8)))  float  v8f;

union FragB16 { v16bf v; uint4 u[2]; };

#define WMMA_BF16(a, b, c) \
  __builtin_amdgcn_wmma_f32_16x16x32_bf16(false, (a), false, (b), (short)0, (c), false, false)

// Problem constants
#define BATCH 4
#define SEQ   4096
#define DIM   1024          // d_model = d_k = d_v
#define BSROWS (BATCH * SEQ)
#define SCALE 0.03125f      // 1/sqrt(1024)

__device__ __forceinline__ unsigned short f2bf(float f) {
  unsigned int u = __float_as_uint(f);
  u += 0x7fffu + ((u >> 16) & 1u);          // round-to-nearest-even
  return (unsigned short)(u >> 16);
}
__device__ __forceinline__ float bf2f(unsigned short h) {
  return __uint_as_float(((unsigned int)h) << 16);
}

// ---------------- 1) f32 -> bf16 convert ----------------
__global__ void cvt_f32_bf16(const float* __restrict__ in,
                             unsigned short* __restrict__ out, int n) {
  int i = blockIdx.x * blockDim.x + threadIdx.x;
  int stride = gridDim.x * blockDim.x;
  for (; i < n; i += stride) out[i] = f2bf(in[i]);
}

// ---------------- 2) C[bs,n] = A[bs,:] . W[n,:] + bias[n] ----------------
// A: [16384][1024] bf16 row-major, W: [1024][1024] bf16 row-major (N,K layout).
// Register-blocked: each wave computes a 64x32 block (4x2 WMMA tiles, 64 acc
// VGPRs -- fits the 256-VGPR window without spilling).
// transV==0 : Out[bs][n] bf16 ; transV==1 : Out[b][n][s] bf16 (V^T)
__global__ void __launch_bounds__(256)
gemm_bias_bf16(const unsigned short* __restrict__ A,
               const unsigned short* __restrict__ W,
               const float* __restrict__ bias,
               unsigned short* __restrict__ Out,
               int transV) {
  int wv   = (blockIdx.x * blockDim.x + threadIdx.x) >> 5; // 0..8191
  int lane = threadIdx.x & 31;
  int lj   = lane & 15;
  int hi   = lane >> 4;
  int mt   = wv >> 5;   // 0..255 : 64-row block of bs
  int nt   = wv & 31;   // 0..31  : 32-col block of n

  const unsigned short* Ap = A + (size_t)(mt * 64 + lj) * DIM;
  const unsigned short* Wp = W + (size_t)(nt * 32 + lj) * DIM;

  v8f acc[4][2];
#pragma unroll
  for (int i = 0; i < 4; i++)
#pragma unroll
    for (int j = 0; j < 2; j++)
#pragma unroll
      for (int r = 0; r < 8; r++) acc[i][j][r] = 0.f;

#pragma unroll 1
  for (int kb = 0; kb < DIM; kb += 32) {
    FragB16 a[4], b[2];
#pragma unroll
    for (int i = 0; i < 4; i++) {
      const unsigned short* p = Ap + (size_t)i * 16 * DIM + kb + hi * 8;
      a[i].u[0] = *(const uint4*)(p);
      a[i].u[1] = *(const uint4*)(p + 16);
    }
#pragma unroll
    for (int j = 0; j < 2; j++) {
      const unsigned short* p = Wp + (size_t)j * 16 * DIM + kb + hi * 8;
      b[j].u[0] = *(const uint4*)(p);
      b[j].u[1] = *(const uint4*)(p + 16);
    }
#pragma unroll
    for (int j = 0; j < 2; j++)
#pragma unroll
      for (int i = 0; i < 4; i++)
        acc[i][j] = WMMA_BF16(a[i].v, b[j].v, acc[i][j]);
  }

  if (!transV) {
#pragma unroll
    for (int j = 0; j < 2; j++) {
      float bn = bias[nt * 32 + j * 16 + lj];
#pragma unroll
      for (int i = 0; i < 4; i++)
#pragma unroll
        for (int r = 0; r < 8; r++) {
          int row = mt * 64 + i * 16 + r + hi * 8;
          Out[(size_t)row * DIM + nt * 32 + j * 16 + lj] = f2bf(acc[i][j][r] + bn);
        }
    }
  } else {
    int b_ = (mt * 64) >> 12;        // batch (64 | 4096, no straddle)
    int s0 = (mt * 64) & (SEQ - 1);  // seq base of 64-row block
#pragma unroll
    for (int j = 0; j < 2; j++) {
      float bn = bias[nt * 32 + j * 16 + lj];
      size_t colbase = ((size_t)b_ * DIM + nt * 32 + j * 16 + lj) * SEQ;
#pragma unroll
      for (int i = 0; i < 4; i++) {
        union { unsigned short h[8]; uint4 u; } st;
#pragma unroll
        for (int r = 0; r < 8; r++) st.h[r] = f2bf(acc[i][j][r] + bn);
        *(uint4*)(Out + colbase + s0 + i * 16 + hi * 8) = st.u;
      }
    }
  }
}

// ---------------- 3) RoPE in place on bf16 Q or K ----------------
__global__ void rope_inplace(unsigned short* __restrict__ Qb,
                             const float* __restrict__ cosv,
                             const float* __restrict__ sinv) {
  int idx = blockIdx.x * blockDim.x + threadIdx.x;   // B*S*512 threads
  int i = idx & 511;
  int s = (idx >> 9) & (SEQ - 1);
  int b = idx >> 21;
  size_t base = ((size_t)b * SEQ + s) * DIM;
  float c  = cosv[s * 512 + i];
  float sn = sinv[s * 512 + i];
  float q0 = bf2f(Qb[base + i]);
  float q1 = bf2f(Qb[base + i + 512]);
  Qb[base + i]       = f2bf(q0 * c - q1 * sn);
  Qb[base + i + 512] = f2bf(q0 * sn + q1 * c);
}

// ---------------- 4) pass 1: row max + row sumexp of causal scores ----------
__global__ void __launch_bounds__(256)
attn_pass1(const unsigned short* __restrict__ Qb,
           const unsigned short* __restrict__ Kb,
           float* __restrict__ mrow, float* __restrict__ lrow) {
  __shared__ __align__(16) unsigned short Qs[16 * DIM]; // 32KB q-tile
  __shared__ float mbuf[8][16];
  __shared__ float lbuf[8][16];

  int b  = blockIdx.x >> 8;
  int qi = blockIdx.x & 255;
  int q0 = qi * 16;
  int tid = threadIdx.x;

  { // stage Q tile (16 x 1024 bf16) into LDS
    const uint4* src = (const uint4*)(Qb + ((size_t)b * SEQ + q0) * DIM);
    uint4* dst = (uint4*)Qs;
    for (int i = tid; i < 2048; i += 256) dst[i] = src[i];
  }
  __syncthreads();

  int w = tid >> 5, lane = tid & 31, lj = lane & 15, hi = lane >> 4;
  const float NEG_INF = -__builtin_inff();
  float mv[8], lv[8];
#pragma unroll
  for (int r = 0; r < 8; r++) { mv[r] = NEG_INF; lv[r] = 0.f; }

  const unsigned short* Qp = Qs + (size_t)lj * DIM;

  // waves process contiguous groups of 4 t-tiles, reusing the Q A-fragment
  for (int tg = w * 4; tg <= qi; tg += 32) {
    const unsigned short* Kp[4];
#pragma unroll
    for (int j = 0; j < 4; j++) {
      int t = (tg + j <= qi) ? (tg + j) : qi;     // clamp: safe, discarded later
      Kp[j] = Kb + ((size_t)b * SEQ + t * 16 + lj) * DIM;
    }
    v8f acc[4];
#pragma unroll
    for (int j = 0; j < 4; j++)
#pragma unroll
      for (int r = 0; r < 8; r++) acc[j][r] = 0.f;

#pragma unroll 1
    for (int kb = 0; kb < DIM; kb += 32) {
      FragB16 a;
      a.u[0] = *(const uint4*)(Qp + kb + hi * 8);
      a.u[1] = *(const uint4*)(Qp + kb + 16 + hi * 8);
#pragma unroll
      for (int j = 0; j < 4; j++) {
        FragB16 bb;
        bb.u[0] = *(const uint4*)(Kp[j] + kb + hi * 8);
        bb.u[1] = *(const uint4*)(Kp[j] + kb + 16 + hi * 8);
        acc[j] = WMMA_BF16(a.v, bb.v, acc[j]);
      }
    }
#pragma unroll
    for (int j = 0; j < 4; j++) {
      int t = tg + j;
      if (t > qi) break;                          // wave-uniform
      bool diag = (t == qi);
#pragma unroll
      for (int r = 0; r < 8; r++) {
        float s = acc[j][r] * SCALE;
        if (diag && (lj > r + hi * 8)) s = NEG_INF;   // causal mask on diagonal
        float nm = fmaxf(mv[r], s);
        if (nm != NEG_INF) {
          lv[r] = lv[r] * __expf(mv[r] - nm) + __expf(s - nm);
          mv[r] = nm;
        }
      }
    }
  }
  // merge across the 16 lanes sharing the same 8 rows
#pragma unroll
  for (int mask = 1; mask <= 8; mask <<= 1) {
#pragma unroll
    for (int r = 0; r < 8; r++) {
      float om = __shfl_xor(mv[r], mask, 32);
      float ol = __shfl_xor(lv[r], mask, 32);
      float nm = fmaxf(mv[r], om);
      float nl = 0.f;
      if (nm != NEG_INF)
        nl = lv[r] * __expf(mv[r] - nm) + ol * __expf(om - nm);
      mv[r] = nm; lv[r] = nl;
    }
  }
  if (lj == 0) {
#pragma unroll
    for (int r = 0; r < 8; r++) { mbuf[w][r + hi * 8] = mv[r]; lbuf[w][r + hi * 8] = lv[r]; }
  }
  __syncthreads();
  if (tid < 16) { // merge across the 8 waves
    float fm = NEG_INF, fl = 0.f;
#pragma unroll
    for (int ww = 0; ww < 8; ww++) {
      float om = mbuf[ww][tid], ol = lbuf[ww][tid];
      float nm = fmaxf(fm, om);
      if (nm != NEG_INF) { fl = fl * __expf(fm - nm) + ol * __expf(om - nm); fm = nm; }
    }
    mrow[(size_t)b * SEQ + q0 + tid] = fm;
    lrow[(size_t)b * SEQ + q0 + tid] = fl;
  }
}

// ---------------- 5) pass 2: z = softmax(scores) @ V ----------------
__global__ void __launch_bounds__(256)
attn_pass2(const unsigned short* __restrict__ Qb,
           const unsigned short* __restrict__ Kb,
           const unsigned short* __restrict__ Vt, // [b][v][t]
           const float* __restrict__ mrow,
           const float* __restrict__ lrow,
           float* __restrict__ Out) {
  __shared__ __align__(16) unsigned short Qs[16 * DIM];  // 32KB
  __shared__ __align__(16) unsigned short Ps[16 * 128];  // 4KB P block

  int b  = blockIdx.x >> 8;
  int qi = blockIdx.x & 255;
  int q0 = qi * 16;
  int tid = threadIdx.x;

  {
    const uint4* src = (const uint4*)(Qb + ((size_t)b * SEQ + q0) * DIM);
    uint4* dst = (uint4*)Qs;
    for (int i = tid; i < 2048; i += 256) dst[i] = src[i];
  }
  __syncthreads();

  int w = tid >> 5, lane = tid & 31, lj = lane & 15, hi = lane >> 4;
  const float NEG_INF = -__builtin_inff();

  float mv[8], li[8];
#pragma unroll
  for (int r = 0; r < 8; r++) {
    int row = q0 + r + hi * 8;
    mv[r] = mrow[(size_t)b * SEQ + row];
    float l = lrow[(size_t)b * SEQ + row];
    li[r] = (l > 0.f) ? (1.0f / l) : 0.f;
  }

  v8f zacc[8];   // wave owns v-cols [w*128, w*128+128)
#pragma unroll
  for (int v = 0; v < 8; v++)
#pragma unroll
    for (int r = 0; r < 8; r++) zacc[v][r] = 0.f;

  int nsteps = (qi >> 3) + 1;                 // 128-wide t-steps
  for (int step = 0; step < nsteps; step++) {
    int tbase = step * 128;
    int t0 = tbase + w * 16;                  // this wave's score sub-tile
    float p[8];
    if (t0 <= q0 + 15) {                      // wave-uniform
      v8f acc = {0.f, 0.f, 0.f, 0.f, 0.f, 0.f, 0.f, 0.f};
      const unsigned short* Qp = Qs + (size_t)lj * DIM;
      const unsigned short* Kp = Kb + ((size_t)b * SEQ + t0 + lj) * DIM;
#pragma unroll 2
      for (int kb = 0; kb < DIM; kb += 32) {
        FragB16 a, bb;
        a.u[0]  = *(const uint4*)(Qp + kb + hi * 8);
        a.u[1]  = *(const uint4*)(Qp + kb + 16 + hi * 8);
        bb.u[0] = *(const uint4*)(Kp + kb + hi * 8);
        bb.u[1] = *(const uint4*)(Kp + kb + 16 + hi * 8);
        acc = WMMA_BF16(a.v, bb.v, acc);
      }
#pragma unroll
      for (int r = 0; r < 8; r++) {
        float s = acc[r] * SCALE;
        if (t0 + lj > q0 + r + hi * 8) s = NEG_INF;   // causal
        p[r] = __expf(s - mv[r]) * li[r];
      }
    } else {
#pragma unroll
      for (int r = 0; r < 8; r++) p[r] = 0.f;
    }
    // C-layout -> row-major P block in LDS (col = w*16 + lj)
#pragma unroll
    for (int r = 0; r < 8; r++)
      Ps[(r + hi * 8) * 128 + w * 16 + lj] = f2bf(p[r]);
    __syncthreads();

    // PV: A = P (16 x 128 over t) hoisted and reused across 8 v-tiles
#pragma unroll
    for (int kc = 0; kc < 4; kc++) {
      FragB16 a;
      const unsigned short* Pp = Ps + (size_t)lj * 128 + kc * 32;
      a.u[0] = *(const uint4*)(Pp + hi * 8);
      a.u[1] = *(const uint4*)(Pp + 16 + hi * 8);
#pragma unroll
      for (int vt = 0; vt < 8; vt++) {
        FragB16 bb;
        const unsigned short* Vp =
            Vt + ((size_t)b * DIM + w * 128 + vt * 16 + lj) * SEQ + tbase + kc * 32;
        bb.u[0] = *(const uint4*)(Vp + hi * 8);
        bb.u[1] = *(const uint4*)(Vp + 16 + hi * 8);
        zacc[vt] = WMMA_BF16(a.v, bb.v, zacc[vt]);
      }
    }
    __syncthreads();   // before P block is overwritten next step
  }

#pragma unroll
  for (int vt = 0; vt < 8; vt++)
#pragma unroll
    for (int r = 0; r < 8; r++) {
      int row = q0 + r + hi * 8;
      Out[((size_t)b * SEQ + row) * DIM + w * 128 + vt * 16 + lj] = zacc[vt][r];
    }
}

// ---------------- host launcher ----------------
extern "C" void kernel_launch(void* const* d_in, const int* in_sizes, int n_in,
                              void* d_out, int out_size, void* d_ws, size_t ws_size,
                              hipStream_t stream) {
  const float* x    = (const float*)d_in[0];
  const float* cosv = (const float*)d_in[1];
  const float* sinv = (const float*)d_in[2];
  const float* Wq   = (const float*)d_in[3];
  const float* bq   = (const float*)d_in[4];
  const float* Wk   = (const float*)d_in[5];
  const float* bk   = (const float*)d_in[6];
  const float* Wv   = (const float*)d_in[7];
  const float* bv   = (const float*)d_in[8];
  float* out = (float*)d_out;

  char* ws = (char*)d_ws;
  size_t off = 0;
  auto carve = [&](size_t bytes) -> char* {
    char* p = ws + off;
    off += (bytes + 255) & ~(size_t)255;
    return p;
  };
  unsigned short* xb  = (unsigned short*)carve((size_t)BSROWS * DIM * 2); // 32MB
  unsigned short* Wqb = (unsigned short*)carve((size_t)DIM * DIM * 2);
  unsigned short* Wkb = (unsigned short*)carve((size_t)DIM * DIM * 2);
  unsigned short* Wvb = (unsigned short*)carve((size_t)DIM * DIM * 2);
  unsigned short* Qbf = (unsigned short*)carve((size_t)BSROWS * DIM * 2); // 32MB
  unsigned short* Kbf = (unsigned short*)carve((size_t)BSROWS * DIM * 2); // 32MB
  unsigned short* Vtb = (unsigned short*)carve((size_t)BSROWS * DIM * 2); // 32MB ([b][v][t])
  float* mrow = (float*)carve((size_t)BSROWS * 4);
  float* lrow = (float*)carve((size_t)BSROWS * 4);
  (void)ws_size; (void)in_sizes; (void)n_in; (void)out_size;

  // 1) converts
  cvt_f32_bf16<<<4096, 256, 0, stream>>>(x, xb, BSROWS * DIM);
  cvt_f32_bf16<<<1024, 256, 0, stream>>>(Wq, Wqb, DIM * DIM);
  cvt_f32_bf16<<<1024, 256, 0, stream>>>(Wk, Wkb, DIM * DIM);
  cvt_f32_bf16<<<1024, 256, 0, stream>>>(Wv, Wvb, DIM * DIM);

  // 2) projections: 8192 wave-blocks of 64x32, 8 waves/block
  gemm_bias_bf16<<<1024, 256, 0, stream>>>(xb, Wqb, bq, Qbf, 0);
  gemm_bias_bf16<<<1024, 256, 0, stream>>>(xb, Wkb, bk, Kbf, 0);
  gemm_bias_bf16<<<1024, 256, 0, stream>>>(xb, Wvb, bv, Vtb, 1); // V transposed

  // 3) RoPE on Q and K (B*S*512 threads each)
  rope_inplace<<<32768, 256, 0, stream>>>(Qbf, cosv, sinv);
  rope_inplace<<<32768, 256, 0, stream>>>(Kbf, cosv, sinv);

  // 4) softmax stats, 5) output — one 16-row q-tile per workgroup
  attn_pass1<<<BATCH * (SEQ / 16), 256, 0, stream>>>(Qbf, Kbf, mrow, lrow);
  attn_pass2<<<BATCH * (SEQ / 16), 256, 0, stream>>>(Qbf, Kbf, Vtb, mrow, lrow, out);
}
